// RnnSpeaker_5935644803364
// MI455X (gfx1250) — matrix operand order repaired and checked by reading, not compiled, not run on hardware
//
#include <hip/hip_runtime.h>
#include <hip/hip_bf16.h>

// ---------------------------------------------------------------------------
// RnnSpeaker for MI455X (gfx1250): fused bf16-WMMA GEMM + argmax GRU decoder.
// message_logits is one-hot, so the [64,32000] logits per step never touch
// memory: WMMA accumulators reduce straight to a per-row argmax atomic.
// h (the A matrix) is staged into LDS once per block via the Tensor Data
// Mover (tensor_load_to_lds + s_wait_tensorcnt), A frags then come from LDS.
// ---------------------------------------------------------------------------

typedef unsigned short ushort_t;
typedef __attribute__((ext_vector_type(16))) __bf16 v16bf;
typedef __attribute__((ext_vector_type(8)))  float  v8f;
typedef __attribute__((ext_vector_type(4)))  unsigned int v4u;
typedef __attribute__((ext_vector_type(8)))  int v8i;
typedef __attribute__((ext_vector_type(4)))  int v4i;

struct alignas(16) U4 { unsigned x, y, z, w; };
union Frag16 { v16bf v; U4 q[2]; };

#define BATCH 64
#define SEQ   32
#define VOCAB 32000
#define EMB   256
#define HID   512
#define EOS_ID 3
#define BOS_ID 1

// float -> bf16, round-to-nearest-even
__device__ __forceinline__ ushort_t f2bf(float f) {
    unsigned u = __float_as_uint(f);
    unsigned r = (u + 0x7FFFu + ((u >> 16) & 1u)) >> 16;
    return (ushort_t)r;
}
// monotone unsigned ordering of float bits (for atomicMax argmax)
__device__ __forceinline__ unsigned f2ord(float f) {
    unsigned u = __float_as_uint(f);
    return (u & 0x80000000u) ? ~u : (u | 0x80000000u);
}

// -------------------------- W_out -> bf16 ----------------------------------
__global__ void wcvt_kernel(const float* __restrict__ W, ushort_t* __restrict__ Wb, int n) {
    int i = blockIdx.x * blockDim.x + threadIdx.x;
    int stride = gridDim.x * blockDim.x;
    for (; i < n; i += stride) Wb[i] = f2bf(W[i]);
}

// -------------------------- zero the one-hot output ------------------------
struct alignas(16) F4 { float a, b, c, d; };
__global__ void zero_kernel(F4* __restrict__ p, long long n4) {
    long long i = (long long)blockIdx.x * blockDim.x + threadIdx.x;
    long long stride = (long long)gridDim.x * blockDim.x;
    F4 z; z.a = z.b = z.c = z.d = 0.0f;
    for (; i < n4; i += stride) p[i] = z;
}

// ---------------- fused logits GEMM (bf16 WMMA) + row argmax ---------------
// grid: 500 blocks (64 vocab cols each), block: 128 threads = 4 waves.
// TDM stages h [64x512 bf16, 64KB] into LDS once; each wave then computes
// 16 vocab columns x all 64 batch rows with 64 v_wmma_f32_16x16x32_bf16.
__global__ __launch_bounds__(128)
void gemm_argmax_kernel(const ushort_t* __restrict__ Wb,   // [V,512] bf16
                        const ushort_t* __restrict__ hb,   // [64,512] bf16
                        const float*    __restrict__ bout, // [V]
                        const float*    __restrict__ gum,  // [64,V] (this step)
                        unsigned long long* __restrict__ amax) // [64]
{
    __shared__ ushort_t hsm[BATCH * HID];    // 64 KB of the 320 KB WGP LDS

    const int lane    = threadIdx.x & 31;
    const int wave    = threadIdx.x >> 5;
    const int nlocal  = lane & 15;           // N within 16-wide tile
    const int hi      = lane >> 4;           // 0: lanes 0-15, 1: lanes 16-31
    const int colBase = blockIdx.x * 64 + wave * 16;
    const int v       = colBase + nlocal;    // vocab column this lane owns

    // ---- TDM: one wave issues a 2-D tensor_load_to_lds for all of h -------
    if (threadIdx.x < 32) {                  // wave-uniform branch (wave 0)
        const unsigned long long ga = (unsigned long long)(size_t)hb;
        const unsigned lds_base = (unsigned)(size_t)(void*)hsm; // addr[31:0] = LDS offset
        // D# group0: count=1 | lds_addr | global_addr[56:0] | type=2
        v4u g0 = { 1u,
                   lds_base,
                   (unsigned)(ga & 0xffffffffu),
                   (unsigned)((ga >> 32) & 0x1ffffffu) | (2u << 30) };
        // D# group1: data_size=1(2B); tensor 512x64; tile 512x64; stride0=512
        v8i g1 = { (int)(1u << 16),          // workgroup_mask=0, data_size=1
                   (int)(512u << 16),        // tensor_dim0 = 512 (bits 79:48)
                   (int)(64u << 16),         // tensor_dim0_hi=0 | tensor_dim1 = 64
                   (int)(512u << 16),        // tensor_dim1_hi=0 | tile_dim0 = 512
                   64,                       // tile_dim1 = 64, tile_dim2 = 0
                   512,                      // tensor_dim0_stride = 512
                   0, 0 };
        v4i z4 = { 0, 0, 0, 0 };
#if defined(__clang_major__) && (__clang_major__ >= 23)
        v8i z8 = { 0, 0, 0, 0, 0, 0, 0, 0 };
        __builtin_amdgcn_tensor_load_to_lds(g0, g1, z4, z4, z8, 0);
#else
        __builtin_amdgcn_tensor_load_to_lds(g0, g1, z4, z4, 0);
#endif
        __builtin_amdgcn_s_wait_tensorcnt(0);
    }
    __syncthreads();

    v8f acc[4] = {};

    for (int k0 = 0; k0 < HID; k0 += 32) {
        // B fragment (32x16 K-major tile of W_out^T): lane holds column v,
        // K = k0 + 16*hi + e (e = 0..15), contiguous in memory.
        Frag16 bf;
        const ushort_t* wp = Wb + (size_t)v * HID + k0 + (hi << 4);
        bf.q[0] = *(const U4*)(wp);
        bf.q[1] = *(const U4*)(wp + 8);
        if (k0 + 32 < HID) __builtin_prefetch((const void*)(wp + 32), 0, 0);

        // A fragments from LDS (16x32 bf16, contiguous K to match B):
        // elems 0-15 -> K=k0+16*hi+e.
#pragma unroll
        for (int mt = 0; mt < 4; ++mt) {
            const int m = mt * 16 + nlocal;
            const ushort_t* hp = hsm + m * HID + k0 + (hi << 4);
            Frag16 af;
            af.q[0] = *(const U4*)(hp);        // ds_load_b128
            af.q[1] = *(const U4*)(hp + 8);    // ds_load_b128
            acc[mt] = __builtin_amdgcn_wmma_f32_16x16x32_bf16(
                false, af.v, false, bf.v, (short)0, acc[mt], false, false);
        }
    }

    // Epilogue: logit + b_out + gumbel, per-batch-row argmax over the 16
    // columns (butterfly shuffle in 16-lane groups), merged via a packed
    // (ordered-float | ~index) u64 atomicMax -> first-max tie rule.
    const float bias = bout[v];
#pragma unroll
    for (int mt = 0; mt < 4; ++mt) {
#pragma unroll
        for (int r = 0; r < 8; ++r) {
            const int b = mt * 16 + r + (hi << 3);   // C/D layout: M = r + 8*hi
            float val = acc[mt][r] + bias + gum[(size_t)b * VOCAB + v];
            int   bi  = v;
            for (int off = 8; off; off >>= 1) {
                float ov = __shfl_xor(val, off, 16);
                int   oi = __shfl_xor(bi,  off, 16);
                if (ov > val || (ov == val && oi < bi)) { val = ov; bi = oi; }
            }
            if (nlocal == 0) {
                unsigned long long key =
                    ((unsigned long long)f2ord(val) << 32) | (unsigned)(~(unsigned)bi);
                atomicMax(&amax[b], key);
            }
        }
    }
}

// ----------------------------- GRU cell ------------------------------------
// One block per batch row; 256 threads, 2 gate-slots each (H=512).
// step == -1: init (x = emb[BOS], h = h_image), also seeds amax = 0.
// step >= 0 : consume argmax -> token id, x = emb_W[idx], update h in place.
__global__ __launch_bounds__(256)
void gru_kernel(const float* __restrict__ embW,
                const float* __restrict__ Wih, const float* __restrict__ Whh,
                const float* __restrict__ bih, const float* __restrict__ bhh,
                const float* __restrict__ h_image,
                float* __restrict__ h, ushort_t* __restrict__ hb,
                unsigned long long* __restrict__ amax,
                int* __restrict__ ids, int step)
{
    __shared__ float xs[EMB];
    __shared__ float hs[HID];
    __shared__ int   s_idx;
    const int b = blockIdx.x;

    if (threadIdx.x == 0) {                  // single reader: no race vs reset
        int idx = BOS_ID;
        if (step >= 0) {
            idx = (int)(~(unsigned)(amax[b] & 0xffffffffu));
            ids[b * SEQ + step] = idx;
        }
        amax[b] = 0ull;                      // reset for next step's atomics
        s_idx = idx;
    }
    __syncthreads();
    const int idx = s_idx;

    const float* hp = (step < 0) ? (h_image + (size_t)b * HID) : (h + (size_t)b * HID);
    for (int i = threadIdx.x; i < EMB; i += blockDim.x) xs[i] = embW[(size_t)idx * EMB + i];
    for (int i = threadIdx.x; i < HID; i += blockDim.x) hs[i] = hp[i];
    __syncthreads();

#pragma unroll
    for (int jj = 0; jj < 2; ++jj) {
        const int j = threadIdx.x + jj * 256;
        float ir = bih[j], iz = bih[j + HID], inn = bih[j + 2 * HID];
        const float* wr = Wih + (size_t)j * EMB;
        const float* wz = Wih + (size_t)(j + HID) * EMB;
        const float* wn = Wih + (size_t)(j + 2 * HID) * EMB;
        for (int e = 0; e < EMB; ++e) {
            const float x = xs[e];
            ir += x * wr[e]; iz += x * wz[e]; inn += x * wn[e];
        }
        float hr = bhh[j], hz = bhh[j + HID], hn = bhh[j + 2 * HID];
        const float* vr = Whh + (size_t)j * HID;
        const float* vz = Whh + (size_t)(j + HID) * HID;
        const float* vn = Whh + (size_t)(j + 2 * HID) * HID;
        for (int e = 0; e < HID; ++e) {
            const float hv = hs[e];
            hr += hv * vr[e]; hz += hv * vz[e]; hn += hv * vn[e];
        }
        const float r = 1.0f / (1.0f + __expf(-(ir + hr)));
        const float z = 1.0f / (1.0f + __expf(-(iz + hz)));
        const float n = tanhf(inn + r * hn);
        const float hnew = (1.0f - z) * n + z * hs[j];
        h [(size_t)b * HID + j] = hnew;
        hb[(size_t)b * HID + j] = f2bf(hnew);
    }
}

// --------------------------- output epilogue -------------------------------
// message_logits[b,t] = one-hot(ids[b,t]) for t < first_eos, else one-hot(0)
// (last step's forced EOS is itself masked to pad by cumsum semantics).
__global__ void epilogue_kernel(const int* __restrict__ ids,
                                float* __restrict__ out,
                                float* __restrict__ ids_out,
                                float* __restrict__ len_out)
{
    const int b = threadIdx.x;
    if (b >= BATCH) return;
    int first = SEQ - 1;                      // labels_mod forces EOS at t=31
    for (int t = 0; t < SEQ - 1; ++t) {
        if (ids[b * SEQ + t] == EOS_ID) { first = t; break; }
    }
    for (int t = 0; t < SEQ; ++t) {
        const int tok = (t < first) ? ids[b * SEQ + t] : 0;
        out[((size_t)(b * SEQ + t)) * VOCAB + tok] = 1.0f;
        ids_out[b * SEQ + t] = (float)ids[b * SEQ + t];
    }
    len_out[b] = (float)(first + 1);
}

// ---------------------------------------------------------------------------
extern "C" void kernel_launch(void* const* d_in, const int* in_sizes, int n_in,
                              void* d_out, int out_size, void* d_ws, size_t ws_size,
                              hipStream_t stream)
{
    (void)in_sizes; (void)n_in; (void)out_size; (void)ws_size;
    const float* h_image = (const float*)d_in[0];   // [64,512]
    const float* gumbel  = (const float*)d_in[1];   // [32,64,32000]
    const float* embW    = (const float*)d_in[2];   // [32000,256]
    const float* Wih     = (const float*)d_in[3];   // [1536,256]
    const float* Whh     = (const float*)d_in[4];   // [1536,512]
    const float* bih     = (const float*)d_in[5];   // [1536]
    const float* bhh     = (const float*)d_in[6];   // [1536]
    const float* Wout    = (const float*)d_in[7];   // [32000,512]
    const float* bout    = (const float*)d_in[8];   // [32000]

    float* out     = (float*)d_out;                              // [64,32,32000]
    float* ids_out = out + (size_t)BATCH * SEQ * VOCAB;          // [64,32]
    float* len_out = ids_out + (size_t)BATCH * SEQ;              // [64]

    // workspace layout (all 256B-aligned)
    char* ws = (char*)d_ws;
    size_t off = 0;
    ushort_t* Wb = (ushort_t*)(ws + off); off += (size_t)VOCAB * HID * sizeof(ushort_t);
    off = (off + 255) & ~(size_t)255;
    float*    h  = (float*)(ws + off);    off += (size_t)BATCH * HID * sizeof(float);
    off = (off + 255) & ~(size_t)255;
    ushort_t* hb = (ushort_t*)(ws + off); off += (size_t)BATCH * HID * sizeof(ushort_t);
    off = (off + 255) & ~(size_t)255;
    unsigned long long* amax = (unsigned long long*)(ws + off); off += BATCH * 8;
    off = (off + 255) & ~(size_t)255;
    int* ids = (int*)(ws + off);

    // 1) W_out -> bf16 (stays resident in 192MB L2 across all 32 steps)
    wcvt_kernel<<<4096, 256, 0, stream>>>(Wout, Wb, VOCAB * HID);

    // 2) initial GRU step (also zeroes argmax atomics)
    gru_kernel<<<BATCH, 256, 0, stream>>>(embW, Wih, Whh, bih, bhh,
                                          h_image, h, hb, amax, ids, -1);

    // 3) zero the one-hot output region
    zero_kernel<<<8192, 256, 0, stream>>>((F4*)out,
                                          (long long)BATCH * SEQ * VOCAB / 4);

    // 4) 32 sequential decode steps: WMMA GEMM+argmax, then GRU update
    for (int s = 0; s < SEQ; ++s) {
        const float* gum_s = gumbel + (size_t)s * BATCH * VOCAB;
        gemm_argmax_kernel<<<VOCAB / 64, 128, 0, stream>>>(Wb, hb, bout, gum_s, amax);
        gru_kernel<<<BATCH, 256, 0, stream>>>(embW, Wih, Whh, bih, bhh,
                                              h_image, h, hb, amax, ids, s);
    }

    // 5) scatter one-hots, ids, lengths
    epilogue_kernel<<<1, 64, 0, stream>>>(ids, out, ids_out, len_out);
}
